// RNN_11828339933262
// MI455X (gfx1250) — matrix operand order repaired and checked by reading, not compiled, not run on hardware
//
#include <hip/hip_runtime.h>
#include <hip/hip_bf16.h>

typedef __attribute__((ext_vector_type(16))) _Float16     v16h;
typedef __attribute__((ext_vector_type(8)))  float        v8f;
typedef __attribute__((ext_vector_type(4)))  unsigned int v4u;
typedef __attribute__((ext_vector_type(8)))  int          v8i;
typedef __attribute__((ext_vector_type(4)))  int          v4i;

#define MROWS  8192   // B*S
#define DDIM   2048   // D_IN = D_STATE = D_OUT
#define NHEADS 32
#define HDIM   64
#define SLEN   2048

// GEMM blocking: 128x128 C-tile per 256-thread block, K-stage of 32,
// double-buffered TDM staging.
#define BM  128
#define BN  128
#define BK  32
#define LDT 36        // LDS row stride in f32: 32 data + 4 TDM-pad dwords -> conflict-free
#define A_TILE (BM * LDT)            // floats per A stage buffer
#define B_TILE (BN * LDT)            // floats per B stage buffer
#define B_BASE (2 * A_TILE)          // offset of B buffers in smem
#define GEMM_LDS_BYTES ((2 * A_TILE + 2 * B_TILE) * 4)   // 73728 B (dynamic LDS)

#if defined(__has_builtin)
#  if __has_builtin(__builtin_amdgcn_tensor_load_to_lds)
#    define HAVE_TDM 1
#  endif
#endif
#ifndef HAVE_TDM
#  define HAVE_TDM 0
#endif

#if HAVE_TDM
// ---------------------------------------------------------------------------
// Issue one TDM 2D tile load: tile_d0=BK f32 contiguous, tile_d1 rows, global
// row stride stride_d0 elements. LDS pad: 4 dwords after every 32 dwords ->
// padded LDS row stride = LDT floats. D# per CDNA5 ISA.
// ---------------------------------------------------------------------------
__device__ __forceinline__ void tdm_load_2d(float* lds_dst, const float* gsrc,
                                            unsigned tensor_d0, unsigned tensor_d1,
                                            unsigned stride_d0, unsigned tile_d1) {
    unsigned long long ga = (unsigned long long)gsrc;
    v4u g0;
    g0[0] = 1u;                                       // count=1, user descriptor
    g0[1] = (unsigned)(unsigned long long)lds_dst;    // lds_addr (offset in LDS)
    g0[2] = (unsigned)ga;                             // global_addr[31:0]
    g0[3] = (unsigned)((ga >> 32) & 0x01FFFFFFu)      // global_addr[56:32]
          | (2u << 30);                               // type = 2 (image)
    v8i g1;
    g1[0] = (int)((2u << 16)                          // data_size = 4B
          | (1u << 20)                                // pad_enable
          | (4u << 22)                                // pad_interval: 32 dwords
          | (3u << 25));                              // pad_amount: 4 dwords
    g1[1] = (int)((tensor_d0 & 0xFFFFu) << 16);                       // dim0 lo16
    g1[2] = (int)(((tensor_d0 >> 16) & 0xFFFFu)
          | ((tensor_d1 & 0xFFFFu) << 16));                           // dim0 hi, dim1 lo
    g1[3] = (int)(((tensor_d1 >> 16) & 0xFFFFu)
          | ((unsigned)BK << 16));                                    // dim1 hi, tile_dim0
    g1[4] = (int)(tile_d1 & 0xFFFFu);                                 // tile_dim1, tile_dim2=0
    g1[5] = (int)stride_d0;                                           // dim0_stride lo32
    g1[6] = 0;                                                        // stride hi bits
    g1[7] = 0;
    v4i z4 = {0, 0, 0, 0};
#  if __clang_major__ >= 23
    v8i z8 = {0, 0, 0, 0, 0, 0, 0, 0};
    __builtin_amdgcn_tensor_load_to_lds(g0, g1, z4, z4, z8, 0);
#  else
    __builtin_amdgcn_tensor_load_to_lds(g0, g1, z4, z4, 0);
#  endif
}
#endif

// ---------------------------------------------------------------------------
// C = A(8192x2048) x W^T (+bias). 256 threads = 8 waves (4x2), each wave
// computes 32x64 = 2x4 WMMA tiles. TDM double-buffers A/B K-stages in LDS:
// stage k+1 DMA overlaps stage k WMMAs; s_wait_tensorcnt 2 gates the oldest
// pair of transfers (TDM completes in-order per wave).
// ---------------------------------------------------------------------------
__global__ void __launch_bounds__(256)
wmma_gemm_kernel(const float* __restrict__ A, const float* __restrict__ W,
                 const float* __restrict__ bias, float* __restrict__ C) {
    extern __shared__ float smem[];

    const int lane = threadIdx.x & 31;
    const int wave = threadIdx.x >> 5;
    const int wm = wave >> 1;              // 0..3  -> 32-row strip
    const int wn = wave & 1;               // 0..1  -> 64-col strip
    const int tm = blockIdx.x * BM;
    const int tn = blockIdx.y * BN;

    v8f acc[2][4] = {};

#if HAVE_TDM
    if (wave == 0) {                       // prologue: stage 0 into buffer 0
        tdm_load_2d(smem,          A + (size_t)tm * DDIM, DDIM, MROWS, DDIM, BM);
        tdm_load_2d(smem + B_BASE, W + (size_t)tn * DDIM, DDIM, DDIM, DDIM, BN);
    }
#endif

    for (int k0 = 0; k0 < DDIM; k0 += BK) {
        const unsigned cur = (unsigned)(k0 / BK) & 1u;
        const float* As = smem + (cur ? A_TILE : 0u);
        const float* Bs = smem + B_BASE + (cur ? B_TILE : 0u);
#if HAVE_TDM
        if (wave == 0) {
            if (k0 + BK < DDIM) {          // issue next stage into other buffer
                float* An = smem + (cur ? 0u : A_TILE);
                float* Bn = smem + B_BASE + (cur ? 0u : B_TILE);
                tdm_load_2d(An, A + (size_t)tm * DDIM + k0 + BK,
                            DDIM, MROWS, DDIM, BM);
                tdm_load_2d(Bn, W + (size_t)tn * DDIM + k0 + BK,
                            DDIM, DDIM, DDIM, BN);
                __builtin_amdgcn_s_wait_tensorcnt(2);  // oldest pair (stage k) done
            } else {
                __builtin_amdgcn_s_wait_tensorcnt(0);
            }
        }
        __syncthreads();                   // current stage visible to all waves
#else
        __syncthreads();                   // previous stage fully consumed
        for (int idx = threadIdx.x; idx < BM * BK; idx += 256) {
            int r = idx >> 5, c = idx & 31;
            smem[r * LDT + c] = A[(size_t)(tm + r) * DDIM + k0 + c];
        }
        for (int idx = threadIdx.x; idx < BN * BK; idx += 256) {
            int r = idx >> 5, c = idx & 31;
            smem[B_BASE + r * LDT + c] = W[(size_t)(tn + r) * DDIM + k0 + c];
        }
        __syncthreads();
        As = smem;
        Bs = smem + B_BASE;
#endif

        // A fragments (16x32 f16): lane<16 -> M=lane&15, K 0..15;
        // lane>=16 -> K 16..31.
        v16h af[2];
#pragma unroll
        for (int i = 0; i < 2; ++i) {
            const float* p = As + (wm * 32 + i * 16 + (lane & 15)) * LDT
                           + ((lane >> 4) << 4);
#pragma unroll
            for (int e = 0; e < 8; ++e) af[i][e]     = (_Float16)p[e];
#pragma unroll
            for (int e = 0; e < 8; ++e) af[i][8 + e] = (_Float16)p[8 + e];
        }
        // B fragments (32x16 f16): lane owns column n=lane&15, one 16-K run.
        v16h bf[4];
#pragma unroll
        for (int j = 0; j < 4; ++j) {
            const float* p = Bs + (wn * 64 + j * 16 + (lane & 15)) * LDT
                           + ((lane >> 4) << 4);
#pragma unroll
            for (int e = 0; e < 16; ++e) bf[j][e] = (_Float16)p[e];
        }
#pragma unroll
        for (int i = 0; i < 2; ++i)
#pragma unroll
            for (int j = 0; j < 4; ++j)
                acc[i][j] = __builtin_amdgcn_wmma_f32_16x16x32_f16(
                    false, af[i], false, bf[j], (short)0, acc[i][j], false, false);

        __syncthreads();                   // stage consumed before TDM reuses buffer
    }

    // C layout: VGPR r -> M = r + 8*(lane>=16), N = lane&15
#pragma unroll
    for (int i = 0; i < 2; ++i) {
#pragma unroll
        for (int j = 0; j < 4; ++j) {
            int col  = tn + wn * 64 + j * 16 + (lane & 15);
            int row0 = tm + wm * 32 + i * 16 + ((lane >> 4) << 3);
            float bv = bias ? bias[col] : 0.0f;
#pragma unroll
            for (int r = 0; r < 8; ++r)
                C[(size_t)(row0 + r) * DDIM + col] = acc[i][j][r] + bv;
        }
    }
}

// ---------------------------------------------------------------------------
// Sequential scan, in-place on h: s_t = tanh(s_{t-1} @ W_n + h_t).
// One block per (b,n); thread k keeps column k of W in 64 VGPRs, s in LDS.
// ---------------------------------------------------------------------------
__global__ void __launch_bounds__(64)
rnn_scan_kernel(float* __restrict__ h, const float* __restrict__ state_weight) {
    __shared__ float sbuf[HDIM];

    int b = blockIdx.x >> 5;
    int n = blockIdx.x & 31;
    int k = threadIdx.x;

    const float* W = state_weight + (size_t)n * HDIM * HDIM + k;
    float wreg[HDIM];
#pragma unroll
    for (int j = 0; j < HDIM; ++j) wreg[j] = W[(size_t)j * HDIM];
    sbuf[k] = 0.0f;
    __syncthreads();

    float* hb = h + (size_t)b * SLEN * DDIM + n * HDIM;
    for (int t = 0; t < SLEN; ++t) {
        if (t + 8 < SLEN)
            __builtin_prefetch(hb + (size_t)(t + 8) * DDIM + k, 0, 0);
        float acc = hb[(size_t)t * DDIM + k];
#pragma unroll
        for (int j = 0; j < HDIM; ++j)
            acc = fmaf(sbuf[j], wreg[j], acc);
        acc = tanhf(acc);
        __syncthreads();                 // all lanes done reading old s
        sbuf[k] = acc;
        hb[(size_t)t * DDIM + k] = acc;  // overwrite h_t with state_t
        __syncthreads();                 // new s visible
    }
}

// ---------------------------------------------------------------------------
// In-place RMSNorm: y[row] *= rsqrt(mean(y^2)+eps) * norm_weight
// ---------------------------------------------------------------------------
__global__ void __launch_bounds__(256)
rms_norm_kernel(float* __restrict__ y, const float* __restrict__ nw) {
    __shared__ float red[8];
    __shared__ float s_scale;
    int row = blockIdx.x;
    float* p = y + (size_t)row * DDIM;

    float ss = 0.0f;
    for (int i = threadIdx.x; i < DDIM; i += 256) {
        float v = p[i];
        ss = fmaf(v, v, ss);
    }
#pragma unroll
    for (int off = 16; off > 0; off >>= 1)
        ss += __shfl_down(ss, off, 32);
    if ((threadIdx.x & 31) == 0) red[threadIdx.x >> 5] = ss;
    __syncthreads();
    if (threadIdx.x == 0) {
        float t = 0.0f;
#pragma unroll
        for (int i = 0; i < 8; ++i) t += red[i];
        s_scale = rsqrtf(t * (1.0f / DDIM) + 1e-6f);
    }
    __syncthreads();
    float sc = s_scale;
    for (int i = threadIdx.x; i < DDIM; i += 256)
        p[i] = p[i] * sc * nw[i];
}

// ---------------------------------------------------------------------------
extern "C" void kernel_launch(void* const* d_in, const int* in_sizes, int n_in,
                              void* d_out, int out_size, void* d_ws, size_t ws_size,
                              hipStream_t stream) {
    const float* x       = (const float*)d_in[0];
    const float* w_in    = (const float*)d_in[1];
    const float* b_in    = (const float*)d_in[2];
    const float* state_w = (const float*)d_in[3];
    const float* norm_w  = (const float*)d_in[4];
    const float* w_out   = (const float*)d_in[5];
    float* out = (float*)d_out;

    float* h = (float*)d_ws;                         // 8192*2048 f32 = 64 MB

    dim3 grid(MROWS / BM, DDIM / BN);                // (64, 16)
    wmma_gemm_kernel<<<grid, 256, GEMM_LDS_BYTES, stream>>>(x, w_in, b_in, h);
    rnn_scan_kernel<<<4 * NHEADS, HDIM, 0, stream>>>(h, state_w);
    rms_norm_kernel<<<MROWS, 256, 0, stream>>>(h, norm_w);
    wmma_gemm_kernel<<<grid, 256, GEMM_LDS_BYTES, stream>>>(h, w_out, nullptr, out);
}
